// BarPooling_29618094473979
// MI455X (gfx1250) — matrix operand order repaired and checked by reading, not compiled, not run on hardware
//
#include <hip/hip_runtime.h>

// ---------------------------------------------------------------------------
// BarPooling on MI455X (gfx1250, wave32, WMMA)
//   stage 1: segment-mean pooling (K=4) + fourier pos-embed -> bf16 H [16384 x 1056]
//   stage 2: W f32 -> bf16
//   stage 3: bf16 WMMA GEMM: out[16384 x 1024] = H @ W^T + b  (f32 accumulate)
//            128x256 block tile, 64x64 per wave, double-buffered LDS,
//            async global->LDS copies (ASYNCcnt) when the toolchain exposes them.
// ---------------------------------------------------------------------------

typedef __attribute__((ext_vector_type(16))) __bf16        v16bf;
typedef __attribute__((ext_vector_type(4)))  __bf16        v4bf;
typedef __attribute__((ext_vector_type(8)))  float         v8f;
typedef __attribute__((ext_vector_type(4)))  float         v4f;
typedef __attribute__((ext_vector_type(4)))  unsigned int  v4u;
typedef __attribute__((ext_vector_type(4)))  int           v4i;

#define D_IN    1024
#define POS_DIM 32
#define H_DIM   1056            // D_IN + POS_DIM
#define MB      2048            // bars per batch row (8192 / 4)
#define ROWS    16384           // B * MB
#define BM      128
#define BN      256
#define BK      32
#define LDS_ROW_U 20            // 80 bytes per LDS tile row (64B data + 16B pad)
#define A_U     (BM * LDS_ROW_U)        // 2560 uints
#define B_U     (BN * LDS_ROW_U)        // 5120 uints
#define BUF_U   (A_U + B_U)             // 7680 uints = 30 KB per buffer
#define KTILES  (H_DIM / BK)            // 33

#if defined(__gfx1250__) && __has_builtin(__builtin_amdgcn_global_load_async_to_lds_b128)
#define HAVE_ASYNC_LDS 1
typedef __attribute__((address_space(1))) v4i gv4i;   // int4 in global AS
typedef __attribute__((address_space(3))) v4i lv4i;   // int4 in LDS AS
#if __has_builtin(__builtin_amdgcn_s_wait_asynccnt)
#define WAIT_ASYNC() __builtin_amdgcn_s_wait_asynccnt(0)
#else
#define WAIT_ASYNC() asm volatile("s_wait_asynccnt 0x0" ::: "memory")
#endif
#else
#define HAVE_ASYNC_LDS 0
#endif

// ------------------------- stage 1: pooling + fourier -----------------------
__global__ __launch_bounds__(256) void pool_ff_kernel(const float* __restrict__ x,
                                                      __bf16* __restrict__ H) {
  const int row = blockIdx.x;            // 0..16383
  const int b   = row >> 11;             // row / 2048
  const int m   = row & (MB - 1);        // bar index
  const int tid = threadIdx.x;

  // mean over 4 consecutive beats (counts are uniformly 4 since M % K == 0)
  const float* src = x + ((size_t)(b * 8192 + m * 4)) * D_IN + tid * 4;
  float s0 = 0.f, s1 = 0.f, s2 = 0.f, s3 = 0.f;
#pragma unroll
  for (int j = 0; j < 4; ++j) {
    const v4f v = *(const v4f*)(src + (size_t)j * D_IN);
    s0 += v.x; s1 += v.y; s2 += v.z; s3 += v.w;
  }
  v4bf o;
  o.x = (__bf16)(s0 * 0.25f);
  o.y = (__bf16)(s1 * 0.25f);
  o.z = (__bf16)(s2 * 0.25f);
  o.w = (__bf16)(s3 * 0.25f);
  *(v4bf*)(H + (size_t)row * H_DIM + tid * 4) = o;

  // fourier positional features: cols 1024..1055
  if (tid < POS_DIM) {
    const int   h    = tid & 15;
    const float pos  = (float)m * (1.0f / 2047.0f);             // m in [0,2047] -> [0,1]
    const float freq = __expf((float)h * 0.46051701859880916f); // log(1000)/15
    const float ang  = pos * freq;
    const float v    = (tid & 16) ? __cosf(ang) : __sinf(ang);
    H[(size_t)row * H_DIM + D_IN + tid] = (__bf16)v;
  }
}

// ------------------------- stage 2: W f32 -> bf16 ---------------------------
__global__ __launch_bounds__(256) void wconv_kernel(const float* __restrict__ W,
                                                    __bf16* __restrict__ Wb) {
  const int i = (blockIdx.x * 256 + threadIdx.x) * 4;   // grid sized exactly
  const v4f v = *(const v4f*)(W + i);
  v4bf o;
  o.x = (__bf16)v.x; o.y = (__bf16)v.y; o.z = (__bf16)v.z; o.w = (__bf16)v.w;
  *(v4bf*)(Wb + i) = o;
}

// ------------------------- stage 3: WMMA GEMM -------------------------------
struct Frag { union { v4u u[2]; v16bf v; }; };

// One BK=32 MMA step for a wave's 64x64 tile. Holds 4 A frags + 1 B frag live.
// A-frag layout (ISA 7.12.2, 16-bit A 16x32): lane L<16 row M=L holds K0..7 and
// K16..23; lane L+16 holds K8..15 and K24..31.  B-frag (32x16): lane L<16 col
// N=L holds K0..7 and K16..23; lane L+16 holds K8..15 and K24..31.
__device__ __forceinline__ void mma_step(const unsigned int* __restrict__ buf,
                                         int wm, int wn, int lrow, int chunk,
                                         v8f (&acc)[4][4]) {
  Frag afr[4];
#pragma unroll
  for (int mi = 0; mi < 4; ++mi) {
    const int r = (wm * 64 + mi * 16 + lrow) * LDS_ROW_U;
    afr[mi].u[0] = *(const v4u*)&buf[r + chunk * 4];
    afr[mi].u[1] = *(const v4u*)&buf[r + 8 + chunk * 4];
  }
#pragma unroll
  for (int ni = 0; ni < 4; ++ni) {
    Frag bfr;
    const int r = A_U + (wn * 64 + ni * 16 + lrow) * LDS_ROW_U;
    bfr.u[0] = *(const v4u*)&buf[r + chunk * 4];
    bfr.u[1] = *(const v4u*)&buf[r + 8 + chunk * 4];
#pragma unroll
    for (int mi = 0; mi < 4; ++mi)
      acc[mi][ni] = __builtin_amdgcn_wmma_f32_16x16x32_bf16(
          false, afr[mi].v, false, bfr.v, (short)0, acc[mi][ni], false, false);
  }
}

__global__ __launch_bounds__(256) void gemm_wmma_kernel(const __bf16* __restrict__ H,
                                                        const __bf16* __restrict__ Wb,
                                                        const float* __restrict__ bias,
                                                        float* __restrict__ out) {
  __shared__ unsigned int lds[2 * BUF_U];   // 60 KB, double buffered

  const int tid   = threadIdx.x;
  const int lane  = tid & 31;
  const int wave  = tid >> 5;
  const int wm    = wave & 1;          // 64-row slab
  const int wn    = wave >> 1;         // 64-col slab
  const int lrow  = lane & 15;
  const int chunk = lane >> 4;

  const int blockN = blockIdx.x;       // 0..3
  const int blockM = blockIdx.y;       // 0..127

  // per-thread copy chunks (16B each): 2 for the A tile (128x32), 4 for B (256x32)
  size_t gA[2]; int lA[2];
#pragma unroll
  for (int i = 0; i < 2; ++i) {
    const int c = tid + i * 256;
    gA[i] = (size_t)(blockM * BM + (c >> 2)) * H_DIM + (c & 3) * 8;
    lA[i] = (c >> 2) * LDS_ROW_U + (c & 3) * 4;
  }
  size_t gB[4]; int lB[4];
#pragma unroll
  for (int i = 0; i < 4; ++i) {
    const int c = tid + i * 256;
    gB[i] = (size_t)(blockN * BN + (c >> 2)) * H_DIM + (c & 3) * 8;
    lB[i] = A_U + (c >> 2) * LDS_ROW_U + (c & 3) * 4;
  }

  v8f acc[4][4];
#pragma unroll
  for (int mi = 0; mi < 4; ++mi)
#pragma unroll
    for (int ni = 0; ni < 4; ++ni)
      acc[mi][ni] = (v8f){0.f, 0.f, 0.f, 0.f, 0.f, 0.f, 0.f, 0.f};

#if HAVE_ASYNC_LDS
  // --- async global->LDS path (ASYNCcnt), one barrier per k-step -----------
  {
    unsigned int* base = lds;
#pragma unroll
    for (int i = 0; i < 2; ++i)
      __builtin_amdgcn_global_load_async_to_lds_b128(
          (gv4i*)(H + gA[i]), (lv4i*)(base + lA[i]), 0, 0);
#pragma unroll
    for (int i = 0; i < 4; ++i)
      __builtin_amdgcn_global_load_async_to_lds_b128(
          (gv4i*)(Wb + gB[i]), (lv4i*)(base + lB[i]), 0, 0);
  }
  WAIT_ASYNC();
  __syncthreads();

  for (int kt = 0; kt < KTILES; ++kt) {
    const int cur = kt & 1;
    if (kt + 1 < KTILES) {
      const int kOff = (kt + 1) * BK;
      unsigned int* base = lds + (cur ^ 1) * BUF_U;
#pragma unroll
      for (int i = 0; i < 2; ++i)
        __builtin_amdgcn_global_load_async_to_lds_b128(
            (gv4i*)(H + gA[i] + kOff), (lv4i*)(base + lA[i]), 0, 0);
#pragma unroll
      for (int i = 0; i < 4; ++i)
        __builtin_amdgcn_global_load_async_to_lds_b128(
            (gv4i*)(Wb + gB[i] + kOff), (lv4i*)(base + lB[i]), 0, 0);
    }
    mma_step(lds + cur * BUF_U, wm, wn, lrow, chunk, acc);
    WAIT_ASYNC();
    __syncthreads();
  }
#else
  // --- fallback: register-staged double buffer, one barrier per k-step -----
  {
    v4u sA[2], sB[4];
#pragma unroll
    for (int i = 0; i < 2; ++i) sA[i] = *(const v4u*)(H + gA[i]);
#pragma unroll
    for (int i = 0; i < 4; ++i) sB[i] = *(const v4u*)(Wb + gB[i]);
#pragma unroll
    for (int i = 0; i < 2; ++i) *(v4u*)&lds[lA[i]] = sA[i];
#pragma unroll
    for (int i = 0; i < 4; ++i) *(v4u*)&lds[lB[i]] = sB[i];
  }
  __syncthreads();

  for (int kt = 0; kt < KTILES; ++kt) {
    const int  cur  = kt & 1;
    const bool more = (kt + 1 < KTILES);
    v4u sA[2], sB[4];
    if (more) {
      const int kOff = (kt + 1) * BK;
#pragma unroll
      for (int i = 0; i < 2; ++i) sA[i] = *(const v4u*)(H + gA[i] + kOff);
#pragma unroll
      for (int i = 0; i < 4; ++i) sB[i] = *(const v4u*)(Wb + gB[i] + kOff);
    }
    mma_step(lds + cur * BUF_U, wm, wn, lrow, chunk, acc);
    if (more) {
      unsigned int* base = lds + (cur ^ 1) * BUF_U;
#pragma unroll
      for (int i = 0; i < 2; ++i) *(v4u*)&base[lA[i]] = sA[i];
#pragma unroll
      for (int i = 0; i < 4; ++i) *(v4u*)&base[lB[i]] = sB[i];
    }
    __syncthreads();
  }
#endif

  // epilogue: C/D layout -> row M = r + 8*(lane>>4), col N = lane&15; fuse bias
#pragma unroll
  for (int mi = 0; mi < 4; ++mi) {
    const int gM0 = blockM * BM + wm * 64 + mi * 16 + chunk * 8;
#pragma unroll
    for (int ni = 0; ni < 4; ++ni) {
      const int   gN = blockN * BN + wn * 64 + ni * 16 + lrow;
      const float bv = bias[gN];
#pragma unroll
      for (int r = 0; r < 8; ++r)
        out[(size_t)(gM0 + r) * D_IN + gN] = acc[mi][ni][r] + bv;
    }
  }
}

// ---------------------------------------------------------------------------
extern "C" void kernel_launch(void* const* d_in, const int* in_sizes, int n_in,
                              void* d_out, int out_size, void* d_ws, size_t ws_size,
                              hipStream_t stream) {
  const float* x    = (const float*)d_in[0];   // [8, 8192, 1024] f32
  const float* W    = (const float*)d_in[1];   // [1024, 1056] f32
  const float* bias = (const float*)d_in[2];   // [1024] f32
  float*       out  = (float*)d_out;           // [8, 2048, 1024] f32

  __bf16* Hb = (__bf16*)d_ws;                                        // 34.6 MB
  __bf16* Wb = (__bf16*)((char*)d_ws + (size_t)ROWS * H_DIM * 2);    // +2.1 MB

  pool_ff_kernel<<<ROWS, 256, 0, stream>>>(x, Hb);
  wconv_kernel<<<(D_IN * H_DIM) / (256 * 4), 256, 0, stream>>>(W, Wb);
  gemm_wmma_kernel<<<dim3(D_IN / BN, ROWS / BM), 256, 0, stream>>>(Hb, Wb, bias, out);
}
